// CardEncoder_79121887527260
// MI455X (gfx1250) — matrix operand order repaired and checked by reading, not compiled, not run on hardware
//
#include <hip/hip_runtime.h>
#include <hip/hip_bf16.h>
#include <cstdint>

typedef _Float16 h16;
typedef __attribute__((ext_vector_type(16))) _Float16 v16h;
typedef __attribute__((ext_vector_type(8)))  float    v8f;
typedef unsigned int u32;
typedef __attribute__((ext_vector_type(4))) u32 u32x4;
typedef __attribute__((ext_vector_type(4))) int i32x4;
typedef __attribute__((ext_vector_type(8))) int i32x8;

#define N_TOT 10240   // B*P
#define L_SEQ 12

#if defined(__has_builtin)
#if __has_builtin(__builtin_amdgcn_tensor_load_to_lds) && __has_builtin(__builtin_amdgcn_s_wait_tensorcnt)
#define XG_ASYNC 1
#endif
#endif

__device__ __forceinline__ float sigf(float x) { return 1.0f / (1.0f + __expf(-x)); }

// Swizzle (row, K) of an A-matrix (16 x K, f16) into the VGPR fragment layout of
// V_WMMA_F32_16X16X32_F16: lane = grp*16 + row, 16 contiguous halves per lane,
// where grp0 covers K%32 in {0..7,16..23} and grp1 covers {8..15,24..31}.
__device__ __forceinline__ int a_swz(int row, int K) {
    int kk  = K >> 5;
    int kw  = K & 31;
    int grp = (kw >> 3) & 1;
    int j   = (kw & 7) + ((kw >> 4) << 3);
    return kk * 512 + (grp * 16 + row) * 16 + j;
}

#ifdef XG_ASYNC
// Tensor Data Mover: 1-D contiguous tile (8-byte elements) global -> LDS.
// D# built per CDNA5 ISA (groups 2/3 disabled: <=2-D tensor).
__device__ __forceinline__ void tdm_load_1d(u32 lds_off, const void* gptr, u32 bytes) {
    u32 elems = bytes >> 3;                    // 8-byte elements
    unsigned long long ga = (unsigned long long)(uintptr_t)gptr;
    u32x4 g0;
    g0[0] = 1u;                                          // count=1, is_restore=0
    g0[1] = lds_off;                                     // lds_addr
    g0[2] = (u32)ga;                                     // global_addr[31:0]
    g0[3] = (u32)((ga >> 32) & 0x01FFFFFFu) | 0x80000000u; // ga[56:32] | type=2
    i32x8 g1;
    g1[0] = 3 << 16;                                     // data_size = 8B
    g1[1] = (int)(elems << 16);                          // tensor_dim0[15:0]
    g1[2] = (int)((elems >> 16) & 0xFFFFu) | (1 << 16);  // tensor_dim0 hi | tensor_dim1=1
    g1[3] = (int)(elems << 16);                          // tile_dim0
    g1[4] = 1;                                           // tile_dim1=1, tile_dim2=0
    g1[5] = (int)elems;                                  // tensor_dim0_stride lo32
    g1[6] = 0;
    g1[7] = 0;
    i32x4 z4 = {0, 0, 0, 0};
#if defined(__clang_major__) && (__clang_major__ >= 23)
    // amdgpu-toolchain (clang-23 / therock-10.0): 6-arg form
    i32x8 z8 = {0, 0, 0, 0, 0, 0, 0, 0};
    __builtin_amdgcn_tensor_load_to_lds(g0, g1, z4, z4, z8, 0);
#else
    // ROCm 7.2 (clang-22): 5-arg form
    __builtin_amdgcn_tensor_load_to_lds(g0, g1, z4, z4, 0);
#endif
}
#endif

// ---------------------------------------------------------------------------
// Pack a [Kreal, Ndim] f32 row-major weight matrix into B-operand tiles:
// tile (kt, nt) = 32x16 block; per lane: N = nt*16 + (lane&15),
// K = kt*32 + (lane>>4)*8 + (j&7) + (j>>3)*16, 16 contiguous halves per lane.
// ---------------------------------------------------------------------------
__global__ void k_packB(const float* __restrict__ W, h16* __restrict__ out,
                        int Kreal, int Ktiles, int Ntiles, int Ndim) {
    int gid  = blockIdx.x * blockDim.x + threadIdx.x;
    int lane = gid & 31;
    int tile = gid >> 5;
    if (tile >= Ktiles * Ntiles) return;
    int kt = tile / Ntiles, nt = tile % Ntiles;
    int grp = lane >> 4;
    int n   = nt * 16 + (lane & 15);
    h16* dst = out + (size_t)tile * 512 + lane * 16;
#pragma unroll
    for (int j = 0; j < 16; ++j) {
        int K = kt * 32 + grp * 8 + (j & 7) + ((j >> 3) << 4);
        float v = (K < Kreal) ? W[(size_t)K * Ndim + n] : 0.0f;
        dst[j] = (h16)v;
    }
}

// ---------------------------------------------------------------------------
// xg[t][n][0:1024] = emb[x[n][t]] @ Wk + b   (f16 out, f32 accum)
// grid (640 row-tiles, 12 timesteps), 128 threads = 4 waves x 16 col-tiles.
// ---------------------------------------------------------------------------
__global__ __launch_bounds__(128)
void k_xg(const int* __restrict__ x, const float* __restrict__ emb,
          const h16* __restrict__ pWk, const float* __restrict__ bias,
          h16* __restrict__ xg) {
    __shared__ __align__(32) h16 sA[10 * 512];   // 16 rows x 320 K, swizzled
    __shared__ int stok[16];
    int nb = blockIdx.x, t = blockIdx.y;
    int tid = threadIdx.x, lane = tid & 31, wave = tid >> 5;
    int nbase = nb * 16;
    if (tid < 16) stok[tid] = x[(size_t)(nbase + tid) * L_SEQ + t];
    __syncthreads();
    for (int idx = tid; idx < 16 * 320; idx += 128) {
        int row = idx / 320, K = idx % 320;
        float v = (K < 300) ? emb[(size_t)stok[row] * 300 + K] : 0.0f;
        sA[a_swz(row, K)] = (h16)v;
    }
    __syncthreads();
    int grp = lane >> 4, nlo = lane & 15;
    for (int ct = 0; ct < 16; ++ct) {
        int colt = wave * 16 + ct;
        int col  = colt * 16 + nlo;
        float bv = bias[col];
        v8f acc = {bv, bv, bv, bv, bv, bv, bv, bv};
#pragma unroll
        for (int kk = 0; kk < 10; ++kk) {
            v16h a = *(const v16h*)(sA + kk * 512 + lane * 16);
            v16h b = *(const v16h*)(pWk + ((size_t)(kk * 64 + colt)) * 512 + lane * 16);
            acc = __builtin_amdgcn_wmma_f32_16x16x32_f16(false, a, false, b,
                                                         (short)0, acc, false, false);
        }
#pragma unroll
        for (int r = 0; r < 8; ++r) {
            int n = nbase + grp * 8 + r;
            xg[((size_t)t * N_TOT + n) * 1024 + col] = (h16)acc[r];
        }
    }
}

// ---------------------------------------------------------------------------
// Bi-LSTM: one workgroup = 16 rows x 1 direction, 12 timesteps fully local.
// xg tiles (32 KB contiguous per step) are double-buffered into LDS by the
// Tensor Data Mover, overlapped with the WMMA pipeline of the previous step.
// ---------------------------------------------------------------------------
__global__ __launch_bounds__(128)
void k_lstm(const h16* __restrict__ xg, const h16* __restrict__ pWr,
            float* __restrict__ hbuf) {
    __shared__ __align__(32) h16 sH[8 * 512];        // 16 x 256 h, swizzled (8 KB)
    __shared__ __align__(16) h16 sXG[2][16 * 1024];  // xg double buffer (2 x 32 KB)
    __shared__ float sZ[16][1024];                   // gate preactivations (64 KB)
    __shared__ float sC[16][256];                    // cell state (16 KB)
    int nb = blockIdx.x, dir = blockIdx.y;
    int tid = threadIdx.x, lane = tid & 31, wave = tid >> 5;
    int nbase = nb * 16;
    int grp = lane >> 4, nlo = lane & 15;
    for (int i = tid; i < 8 * 512; i += 128) sH[i] = (h16)0.0f;
    for (int i = tid; i < 16 * 256; i += 128) sC[i >> 8][i & 255] = 0.0f;
#ifdef XG_ASYNC
    {   // prime buffer 0 with the first timestep
        int t0 = dir ? (L_SEQ - 1) : 0;
        if (wave == 0) {
            tdm_load_1d((u32)(uintptr_t)(void*)&sXG[0][0],
                        xg + ((size_t)t0 * N_TOT + nbase) * 1024, 16 * 1024 * 2);
            __builtin_amdgcn_s_wait_tensorcnt(0);
        }
    }
#endif
    __syncthreads();
    int cur = 0;
    for (int step = 0; step < L_SEQ; ++step) {
#ifdef XG_ASYNC
        if (wave == 0 && step + 1 < L_SEQ) {         // async prefetch of next step
            int tn = dir ? (L_SEQ - 2 - step) : (step + 1);
            tdm_load_1d((u32)(uintptr_t)(void*)&sXG[cur ^ 1][0],
                        xg + ((size_t)tn * N_TOT + nbase) * 1024, 16 * 1024 * 2);
        }
#else
        {   // fallback: synchronous cooperative stage of this step's xg
            int t = dir ? (L_SEQ - 1 - step) : step;
            const h16* src = xg + ((size_t)t * N_TOT + nbase) * 1024;
            for (int i = tid; i < 16 * 1024; i += 128) sXG[cur][i] = src[i];
            __syncthreads();
        }
#endif
        v16h a[8];
#pragma unroll
        for (int kk = 0; kk < 8; ++kk)
            a[kk] = *(const v16h*)(sH + kk * 512 + lane * 16);
        for (int ct = 0; ct < 16; ++ct) {
            int colt = wave * 16 + ct;
            int col  = colt * 16 + nlo;
            v8f acc;
#pragma unroll
            for (int r = 0; r < 8; ++r)
                acc[r] = (float)sXG[cur][(grp * 8 + r) * 1024 + col];
#pragma unroll
            for (int kk = 0; kk < 8; ++kk) {
                v16h b = *(const v16h*)(pWr + ((size_t)(kk * 64 + colt)) * 512 + lane * 16);
                acc = __builtin_amdgcn_wmma_f32_16x16x32_f16(false, a[kk], false, b,
                                                             (short)0, acc, false, false);
            }
#pragma unroll
            for (int r = 0; r < 8; ++r) sZ[grp * 8 + r][col] = acc[r];
        }
        __syncthreads();
        for (int idx = tid; idx < 16 * 256; idx += 128) {
            int row = idx >> 8, c0 = idx & 255;
            float zi = sZ[row][c0];
            float zf = sZ[row][256 + c0];
            float zg = sZ[row][512 + c0];
            float zo = sZ[row][768 + c0];
            float c = sigf(zf) * sC[row][c0] + sigf(zi) * tanhf(zg);
            sC[row][c0] = c;
            sH[a_swz(row, c0)] = (h16)(sigf(zo) * tanhf(c));
        }
#ifdef XG_ASYNC
        if (wave == 0) __builtin_amdgcn_s_wait_tensorcnt(0);
        __syncthreads();   // publish next xg buffer + new h to all waves
        cur ^= 1;
#else
        __syncthreads();
#endif
    }
    for (int idx = tid; idx < 16 * 256; idx += 128) {
        int row = idx >> 8, c0 = idx & 255;
        hbuf[(size_t)(nbase + row) * 512 + dir * 256 + c0] = (float)sH[a_swz(row, c0)];
    }
}

// ---------------------------------------------------------------------------
// dense = tanh(state @ Wd + bd) : [10240,512] x [512,256]
// ---------------------------------------------------------------------------
__global__ __launch_bounds__(128)
void k_dense(const float* __restrict__ hbuf, const h16* __restrict__ pWd,
             const float* __restrict__ bd, float* __restrict__ dense) {
    __shared__ __align__(32) h16 sA[16 * 512];   // 16 rows x 512 K, swizzled
    int nb = blockIdx.x;
    int tid = threadIdx.x, lane = tid & 31, wave = tid >> 5;
    int nbase = nb * 16;
    for (int idx = tid; idx < 16 * 512; idx += 128) {
        int row = idx >> 9, K = idx & 511;
        sA[a_swz(row, K)] = (h16)hbuf[(size_t)(nbase + row) * 512 + K];
    }
    __syncthreads();
    int grp = lane >> 4, nlo = lane & 15;
    for (int ct = 0; ct < 4; ++ct) {
        int colt = wave * 4 + ct;
        int col  = colt * 16 + nlo;
        float bv = bd[col];
        v8f acc = {bv, bv, bv, bv, bv, bv, bv, bv};
#pragma unroll
        for (int kk = 0; kk < 16; ++kk) {
            v16h a = *(const v16h*)(sA + kk * 512 + lane * 16);
            v16h b = *(const v16h*)(pWd + ((size_t)(kk * 16 + colt)) * 512 + lane * 16);
            acc = __builtin_amdgcn_wmma_f32_16x16x32_f16(false, a, false, b,
                                                         (short)0, acc, false, false);
        }
#pragma unroll
        for (int r = 0; r < 8; ++r) {
            int n = nbase + grp * 8 + r;
            dense[(size_t)n * 256 + col] = tanhf(acc[r]);
        }
    }
}

// ---------------------------------------------------------------------------
// Per-card attention over 40 paths: scores, softmax, weighted sum.
// ---------------------------------------------------------------------------
__global__ __launch_bounds__(256)
void k_attn(const float* __restrict__ dense, const float* __restrict__ att,
            float* __restrict__ out) {
    __shared__ float satt[256];
    __shared__ float sred[256];
    __shared__ float sw[40];
    int b = blockIdx.x, tid = threadIdx.x;
    satt[tid] = att[tid];
    __syncthreads();
    for (int p = 0; p < 40; ++p) {
        sred[tid] = dense[((size_t)b * 40 + p) * 256 + tid] * satt[tid];
        __syncthreads();
        for (int s = 128; s > 0; s >>= 1) {
            if (tid < s) sred[tid] += sred[tid + s];
            __syncthreads();
        }
        if (tid == 0) sw[p] = sred[0];
        __syncthreads();
    }
    if (tid == 0) {
        float m = sw[0];
        for (int p = 1; p < 40; ++p) m = fmaxf(m, sw[p]);
        float s = 0.0f;
        for (int p = 0; p < 40; ++p) { sw[p] = __expf(sw[p] - m); s += sw[p]; }
        float inv = 1.0f / s;
        for (int p = 0; p < 40; ++p) sw[p] *= inv;
    }
    __syncthreads();
    float o = 0.0f;
    for (int p = 0; p < 40; ++p)
        o += sw[p] * dense[((size_t)b * 40 + p) * 256 + tid];
    out[(size_t)b * 256 + tid] = o;
}

// ---------------------------------------------------------------------------
extern "C" void kernel_launch(void* const* d_in, const int* in_sizes, int n_in,
                              void* d_out, int out_size, void* d_ws, size_t ws_size,
                              hipStream_t stream) {
    (void)in_sizes; (void)n_in; (void)out_size; (void)ws_size;
    const int*   x   = (const int*)d_in[0];
    const float* emb = (const float*)d_in[1];
    const float* Wk  = (const float*)d_in[2];
    const float* Wr  = (const float*)d_in[3];
    const float* b   = (const float*)d_in[4];
    const float* Wd  = (const float*)d_in[5];
    const float* bd  = (const float*)d_in[6];
    const float* att = (const float*)d_in[7];
    float* out = (float*)d_out;

    char* ws = (char*)d_ws;
    size_t off = 0;
    auto take = [&](size_t bytes) -> char* {
        char* p = ws + off;
        off += (bytes + 255) & ~(size_t)255;
        return p;
    };
    h16*   pWk   = (h16*)take((size_t)10 * 64 * 512 * 2);           // Wk packed
    h16*   pWr   = (h16*)take((size_t)8 * 64 * 512 * 2);            // Wr packed
    h16*   pWd   = (h16*)take((size_t)16 * 16 * 512 * 2);           // Wd packed
    h16*   xg    = (h16*)take((size_t)L_SEQ * N_TOT * 1024 * 2);    // input gates, f16
    float* hbuf  = (float*)take((size_t)N_TOT * 512 * 4);           // [h_fw | h_bw]
    float* dense = (float*)take((size_t)N_TOT * 256 * 4);

    { int tiles = 10 * 64;
      k_packB<<<dim3((tiles * 32 + 127) / 128), 128, 0, stream>>>(Wk, pWk, 300, 10, 64, 1024); }
    { int tiles = 8 * 64;
      k_packB<<<dim3((tiles * 32 + 127) / 128), 128, 0, stream>>>(Wr, pWr, 256, 8, 64, 1024); }
    { int tiles = 16 * 16;
      k_packB<<<dim3((tiles * 32 + 127) / 128), 128, 0, stream>>>(Wd, pWd, 512, 16, 16, 256); }

    k_xg  <<<dim3(640, L_SEQ), 128, 0, stream>>>(x, emb, pWk, b, xg);
    k_lstm<<<dim3(640, 2),     128, 0, stream>>>(xg, pWr, hbuf);
    k_dense<<<640, 128, 0, stream>>>(hbuf, pWd, bd, dense);
    k_attn <<<256, 256, 0, stream>>>(dense, att, out);
}
